// MeshGraphNets_14147622273438
// MI455X (gfx1250) — matrix operand order repaired and checked by reading, not compiled, not run on hardware
//
#include <hip/hip_runtime.h>
#include <hip/hip_bf16.h>

// ---------------------------------------------------------------------------
// MeshGraphNets forward for gfx1250 (MI455X): templated WMMA f16 GEMMs with
// LDS-staged weight tiles, batched B-fragment loads (immediate DS offsets),
// and fused SiLU / LayerNorm / residual / scatter-add epilogues.
// ---------------------------------------------------------------------------

typedef _Float16 half_t;
typedef __attribute__((ext_vector_type(16))) _Float16 v16h;
typedef __attribute__((ext_vector_type(8)))  _Float16 v8h;
typedef __attribute__((ext_vector_type(4)))  _Float16 v4h;
typedef __attribute__((ext_vector_type(8)))  float    v8f;
typedef __attribute__((ext_vector_type(4)))  float    v4f;

#define NN 30000
#define NE 240000
#define L  128

// --- weight transpose+pad: W[K,N] f32 -> Wt[N,K32] f16 (zero-padded K) -----
__global__ __launch_bounds__(256)
void wconv_kernel(const float* __restrict__ w, half_t* __restrict__ wt,
                  int K, int N, int K32) {
    int i = blockIdx.x * 256 + threadIdx.x;
    int total = N * K32;
    if (i >= total) return;
    int n = i / K32, k = i % K32;
    wt[(size_t)n * K32 + k] = (k < K) ? (half_t)w[(size_t)k * N + n] : (half_t)0.f;
}

__global__ __launch_bounds__(256)
void zero_f32_kernel(float* __restrict__ p, int n) {
    int i = blockIdx.x * 256 + threadIdx.x;
    if (i < n) p[i] = 0.f;
}

// --- fused MLP layer: Y[M,128] = epi( stage(X) @ Wt + b ) ------------------
// INK: 0 = f32 plain [M,K_real] (scalar staged), 1 = f16 plain [M,128]
//      (A direct from global, no staging), 2 = edge concat [h[src]|h[dst]|e],
//      3 = node concat [h|agg]
// EPI: 0 = SiLU -> f16 out, 1 = LayerNorm(g,bt) (+resid) -> f32 out,
//          optionally atomic-scatter LN value into atomic_out[aidx[row]]
template <int K32, int INK, int EPI>
__global__ __launch_bounds__(128)
void mlp_gemm_kernel(int M, int K_real,
                     const float* __restrict__ inA_f32,
                     const half_t* __restrict__ inA_f16,
                     const float* __restrict__ h_in,
                     const float* __restrict__ e_in,
                     const float* __restrict__ agg_in,
                     const int* __restrict__ srcIdx,
                     const int* __restrict__ dstIdx,
                     const half_t* __restrict__ Wt,
                     const float* __restrict__ bias,
                     const float* __restrict__ gamma,
                     const float* __restrict__ beta,
                     const float* __restrict__ resid,
                     float* __restrict__ atomic_out,
                     const int* __restrict__ aidx,
                     half_t* __restrict__ out_f16,
                     float* __restrict__ out_f32) {
    constexpr int CK     = (K32 < 128) ? K32 : 128;   // W chunk K-extent
    constexpr int NCHUNK = K32 / CK;
    constexpr int NKK    = CK / 32;
    constexpr bool A_GLOB = (INK == 1);

    extern __shared__ __align__(16) char smem_raw[];
    half_t* ldsA = (half_t*)smem_raw;                 // [64][K32] if staged
    half_t* ldsB = A_GLOB ? (half_t*)smem_raw
                          : (half_t*)(smem_raw + (size_t)64 * K32 * 2);

    const int tile = blockIdx.x * 64;
    const int tid  = threadIdx.x;

    // ---- cooperative stage of A into LDS ------------------------------------
    if (INK == 0) {
        // small/odd-K scalar path with K_real guard
        const int total = 64 * K32;
        for (int idx = tid; idx < total; idx += 128) {
            int r = idx / K32, c = idx - r * K32;
            int row = tile + r;
            float v = 0.f;
            if (row < M && c < K_real) v = inA_f32[(size_t)row * K_real + c];
            ldsA[idx] = (half_t)v;
        }
    } else if (INK == 2 || INK == 3) {
        // concat modes: 2 threads/row, hoisted gather indices, float4 moves
        const int row_l = tid >> 1;                   // 0..63
        const int row   = tile + row_l;
        constexpr int CHALF = K32 / 2;                // 128 or 192
        const int cbeg = (tid & 1) * CHALF;
        if (row < M) {
            const float* seg0;
            const float* seg1;
            const float* seg2 = nullptr;
            if (INK == 2) {
                seg0 = h_in + (size_t)srcIdx[row] * L;
                seg1 = h_in + (size_t)dstIdx[row] * L;
                seg2 = e_in + (size_t)row * L;
            } else {
                seg0 = h_in + (size_t)row * L;
                seg1 = agg_in + (size_t)row * L;
            }
#pragma unroll
            for (int j = 0; j < CHALF / 4; ++j) {
                int c = cbeg + j * 4;
                const float* sp = (c < L)     ? seg0 + c
                                : (c < 2 * L) ? seg1 + (c - L)
                                              : seg2 + (c - 2 * L);
                v4f v = *(const v4f*)sp;
                v4h hv;
                hv[0] = (half_t)v[0]; hv[1] = (half_t)v[1];
                hv[2] = (half_t)v[2]; hv[3] = (half_t)v[3];
                *(v4h*)&ldsA[row_l * K32 + c] = hv;
            }
        } else {
#pragma unroll
            for (int j = 0; j < CHALF / 4; ++j)
                *(v4h*)&ldsA[row_l * K32 + cbeg + j * 4] = (v4h)(half_t)0.f;
        }
    }

    // ---- per-wave 16x128 strip via 8x WMMA 16x16x32 f16 ----
    const int wave = tid >> 5;
    const int lane = tid & 31;
    const int hsel = lane >> 4;
    const int l16  = lane & 15;
    const int rowL = wave * 16 + l16;    // A-frag row within tile
    const int kb   = hsel ? 8 : 0;       // A halves: kb..kb+7, kb+16..kb+23
    int rowA = tile + rowL;              // global A row (direct-load mode)
    if (rowA >= M) rowA = M - 1;         // clamp (stores are guarded)

    v8f acc[8];
#pragma unroll
    for (int t = 0; t < 8; ++t)
#pragma unroll
        for (int r = 0; r < 8; ++r) acc[t][r] = 0.f;

    for (int ch = 0; ch < NCHUNK; ++ch) {
        __syncthreads();  // ldsB reuse fence (also fences A staging, 1st iter)

        // stage W chunk [128 x CK] into LDS (already f16, vector copies)
        constexpr int CNT8 = (128 * CK) / 8;
        constexpr int CK8  = CK / 8;
#pragma unroll 4
        for (int idx = tid; idx < CNT8; idx += 128) {
            int n  = idx / CK8;
            int k8 = idx - n * CK8;
            *(v8h*)&ldsB[n * CK + k8 * 8] =
                *(const v8h*)&Wt[(size_t)n * K32 + ch * 128 + k8 * 8];
        }
        if (NCHUNK > 1 && ch + 1 < NCHUNK)  // prefetch next weight chunk
            __builtin_prefetch(&Wt[(size_t)tid * K32 + (ch + 1) * 128], 0, 0);
        __syncthreads();

#pragma unroll
        for (int kkl = 0; kkl < NKK; ++kkl) {
            const int kk = ch * 4 + kkl;           // global k-step
            union { v16h v; v8h h8[2]; } a;
            if (A_GLOB) {
                const half_t* ap = inA_f16 + (size_t)rowA * L + kk * 32 + kb;
                a.h8[0] = *(const v8h*)(ap);
                a.h8[1] = *(const v8h*)(ap + 16);
            } else {
                const half_t* ap = &ldsA[rowL * K32 + kk * 32 + kb];
                a.h8[0] = *(const v8h*)(ap);
                a.h8[1] = *(const v8h*)(ap + 16);
            }
            // batch all 8 B fragments: same base, immediate DS offsets
            const half_t* bp = ldsB + l16 * CK + kkl * 32 + (hsel ? 8 : 0);
            v16h bf[8];
#pragma unroll
            for (int t = 0; t < 8; ++t) {
                v8h b0 = *(const v8h*)(bp + t * 16 * CK);
                v8h b1 = *(const v8h*)(bp + t * 16 * CK + 16);
                bf[t] = __builtin_shufflevector(b0, b1, 0, 1, 2, 3, 4, 5, 6, 7, 8, 9, 10, 11, 12, 13, 14, 15);
            }
#pragma unroll
            for (int t = 0; t < 8; ++t)
                acc[t] = __builtin_amdgcn_wmma_f32_16x16x32_f16(
                    false, a.v, false, bf[t], (short)0, acc[t], false, false);
        }
    }

    // ---- bias ----
#pragma unroll
    for (int t = 0; t < 8; ++t) {
        float bv = bias[t * 16 + l16];
#pragma unroll
        for (int r = 0; r < 8; ++r) acc[t][r] += bv;
    }

    const int row_base = tile + wave * 16 + hsel * 8;   // + r = global row

    if (EPI == 0) {
        // SiLU -> f16
#pragma unroll
        for (int r = 0; r < 8; ++r) {
            int rowg = row_base + r;
            if (rowg >= M) continue;
#pragma unroll
            for (int t = 0; t < 8; ++t) {
                int col = t * 16 + l16;
                float v = acc[t][r];
                float s = v / (1.f + __expf(-v));
                out_f16[(size_t)rowg * L + col] = (half_t)s;
            }
        }
    } else {
        // LayerNorm over 128 cols: a row lives across 8 reg-tiles x 16 lanes
#pragma unroll
        for (int r = 0; r < 8; ++r) {
            float s1 = 0.f, s2 = 0.f;
#pragma unroll
            for (int t = 0; t < 8; ++t) { float v = acc[t][r]; s1 += v; s2 += v * v; }
            for (int m = 1; m < 16; m <<= 1) {
                s1 += __shfl_xor(s1, m, 32);
                s2 += __shfl_xor(s2, m, 32);
            }
            float mean = s1 * (1.f / 128.f);
            float var  = s2 * (1.f / 128.f) - mean * mean;
            float rs   = rsqrtf(var + 1e-5f);
            int rowg = row_base + r;
            bool ok = rowg < M;
            int dsti = (ok && aidx) ? aidx[rowg] : 0;
#pragma unroll
            for (int t = 0; t < 8; ++t) {
                int col = t * 16 + l16;
                float y = (acc[t][r] - mean) * rs * gamma[col] + beta[col];
                if (ok) {
                    float o = y;
                    if (resid) o += resid[(size_t)rowg * L + col];
                    out_f32[(size_t)rowg * L + col] = o;
                    if (atomic_out)
                        atomicAdd(&atomic_out[(size_t)dsti * L + col], y);
                }
            }
        }
    }
}

// --- decoder final layer: [M,128] f16 @ w3[128,3] + b3 -> out[M,3] f32 -----
__global__ __launch_bounds__(256)
void dec_final_kernel(const half_t* __restrict__ hin,
                      const float* __restrict__ w3,
                      const float* __restrict__ b3,
                      float* __restrict__ out, int M) {
    int i = blockIdx.x * 256 + threadIdx.x;
    if (i >= M) return;
    float a0 = b3[0], a1 = b3[1], a2 = b3[2];
    const half_t* hp = hin + (size_t)i * L;
#pragma unroll 4
    for (int k = 0; k < L; ++k) {
        float hv = (float)hp[k];
        a0 += hv * w3[k * 3 + 0];
        a1 += hv * w3[k * 3 + 1];
        a2 += hv * w3[k * 3 + 2];
    }
    out[(size_t)i * 3 + 0] = a0;
    out[(size_t)i * 3 + 1] = a1;
    out[(size_t)i * 3 + 2] = a2;
}

// ---------------------------------------------------------------------------
extern "C" void kernel_launch(void* const* d_in, const int* in_sizes, int n_in,
                              void* d_out, int out_size, void* d_ws, size_t ws_size,
                              hipStream_t stream) {
    (void)in_sizes; (void)n_in; (void)out_size; (void)ws_size;

    auto F = [&](int i) { return (const float*)d_in[i]; };
    const float* x     = F(0);                 // [NN,11]
    const float* eat   = F(1);                 // [NE,3]
    const int*   eidx  = (const int*)d_in[2];  // [2,NE]
    const int*   src   = eidx;
    const int*   dst   = eidx + NE;

    // workspace carve-out
    char* ws = (char*)d_ws;
    size_t off = 0;
    auto alloc = [&](size_t bytes) { size_t o = off; off = (off + bytes + 255) & ~(size_t)255; return o; };
    float*  h    = (float*) (ws + alloc((size_t)NN * L * 4));
    float*  e    = (float*) (ws + alloc((size_t)NE * L * 4));
    float*  agg  = (float*) (ws + alloc((size_t)NN * L * 4));
    half_t* hid  = (half_t*)(ws + alloc((size_t)NE * L * 2));
    auto wtbuf = [&](size_t halves) { return (half_t*)(ws + alloc(halves * 2)); };

    half_t* wt_en1 = wtbuf((size_t)L * 32);
    half_t* wt_en2 = wtbuf((size_t)L * L);
    half_t* wt_en3 = wtbuf((size_t)L * L);
    half_t* wt_ee1 = wtbuf((size_t)L * 32);
    half_t* wt_ee2 = wtbuf((size_t)L * L);
    half_t* wt_ee3 = wtbuf((size_t)L * L);
    half_t* wt_be1[3]; half_t* wt_be2[3]; half_t* wt_be3[3];
    half_t* wt_bn1[3]; half_t* wt_bn2[3]; half_t* wt_bn3[3];
    for (int i = 0; i < 3; ++i) {
        wt_be1[i] = wtbuf((size_t)L * 384);
        wt_be2[i] = wtbuf((size_t)L * L);
        wt_be3[i] = wtbuf((size_t)L * L);
        wt_bn1[i] = wtbuf((size_t)L * 256);
        wt_bn2[i] = wtbuf((size_t)L * L);
        wt_bn3[i] = wtbuf((size_t)L * L);
    }
    half_t* wt_d1 = wtbuf((size_t)L * L);
    half_t* wt_d2 = wtbuf((size_t)L * L);

    auto conv = [&](const float* w, half_t* wt, int K, int N, int K32) {
        int total = N * K32;
        hipLaunchKernelGGL(wconv_kernel, dim3((total + 255) / 256), dim3(256), 0, stream,
                           w, wt, K, N, K32);
    };

    // input leaves (setup_inputs order):
    // 3..10 enc_node (w1,b1,w2,b2,w3,b3,g,bt) | 11..18 enc_edge
    // 19..26 blk_edge [3] | 27..34 blk_node [3] | 35..40 dec
    conv(F(3),  wt_en1, 11, L, 32);
    conv(F(5),  wt_en2, L,  L, L);
    conv(F(7),  wt_en3, L,  L, L);
    conv(F(11), wt_ee1, 3,  L, 32);
    conv(F(13), wt_ee2, L,  L, L);
    conv(F(15), wt_ee3, L,  L, L);
    for (int i = 0; i < 3; ++i) {
        conv(F(19) + (size_t)i * 384 * L, wt_be1[i], 384, L, 384);
        conv(F(21) + (size_t)i * L * L,   wt_be2[i], L,   L, L);
        conv(F(23) + (size_t)i * L * L,   wt_be3[i], L,   L, L);
        conv(F(27) + (size_t)i * 256 * L, wt_bn1[i], 256, L, 256);
        conv(F(29) + (size_t)i * L * L,   wt_bn2[i], L,   L, L);
        conv(F(31) + (size_t)i * L * L,   wt_bn3[i], L,   L, L);
    }
    conv(F(35), wt_d1, L, L, L);
    conv(F(37), wt_d2, L, L, L);

#define GEMM(K32V, INKV, EPIV, M, K_real, inf32, inf16, h_in, e_in, agg_in,    \
             Wt, bias, g, bt, resid, atomic_out, ai, of16, of32)               \
    do {                                                                       \
        size_t shA = (INKV == 1) ? 0 : (size_t)64 * (K32V)*2;                  \
        size_t shB = (size_t)128 * (((K32V) < 128) ? (K32V) : 128) * 2;        \
        hipLaunchKernelGGL((mlp_gemm_kernel<K32V, INKV, EPIV>),                \
                           dim3(((M) + 63) / 64), dim3(128), shA + shB,        \
                           stream, (M), (K_real), (inf32), (inf16), (h_in),    \
                           (e_in), (agg_in), src, dst, (Wt), (bias), (g),      \
                           (bt), (resid), (atomic_out), (ai), (of16), (of32)); \
    } while (0)

    const float*  nf  = nullptr;   // null const-f32 input
    const half_t* nh  = nullptr;   // null const-f16 input
    const int*    ni  = nullptr;   // null index input
    float*        nfw = nullptr;   // null f32 output
    half_t*       nhw = nullptr;   // null f16 output

    // ---- encoder: nodes ----
    GEMM(32, 0, 0, NN, 11, x, nh, nf, nf, nf, wt_en1, F(4), nf, nf, nf, nfw, ni, hid, nfw);
    GEMM(128, 1, 0, NN, L, nf, hid, nf, nf, nf, wt_en2, F(6), nf, nf, nf, nfw, ni, hid, nfw);
    GEMM(128, 1, 1, NN, L, nf, hid, nf, nf, nf, wt_en3, F(8), F(9), F(10), nf, nfw, ni, nhw, h);
    // ---- encoder: edges ----
    GEMM(32, 0, 0, NE, 3, eat, nh, nf, nf, nf, wt_ee1, F(12), nf, nf, nf, nfw, ni, hid, nfw);
    GEMM(128, 1, 0, NE, L, nf, hid, nf, nf, nf, wt_ee2, F(14), nf, nf, nf, nfw, ni, hid, nfw);
    GEMM(128, 1, 1, NE, L, nf, hid, nf, nf, nf, wt_ee3, F(16), F(17), F(18), nf, nfw, ni, nhw, e);

    // ---- message-passing blocks ----
    for (int i = 0; i < 3; ++i) {
        const float* be_b1 = F(20) + (size_t)i * L;
        const float* be_b2 = F(22) + (size_t)i * L;
        const float* be_b3 = F(24) + (size_t)i * L;
        const float* be_g  = F(25) + (size_t)i * L;
        const float* be_bt = F(26) + (size_t)i * L;
        const float* bn_b1 = F(28) + (size_t)i * L;
        const float* bn_b2 = F(30) + (size_t)i * L;
        const float* bn_b3 = F(32) + (size_t)i * L;
        const float* bn_g  = F(33) + (size_t)i * L;
        const float* bn_bt = F(34) + (size_t)i * L;

        // EdgeBlock: msg = MLP([h[src], h[dst], e])
        GEMM(384, 2, 0, NE, 384, nf, nh, h, e, nf, wt_be1[i], be_b1, nf, nf, nf, nfw, ni, hid, nfw);
        GEMM(128, 1, 0, NE, L, nf, hid, nf, nf, nf, wt_be2[i], be_b2, nf, nf, nf, nfw, ni, hid, nfw);
        // zero agg, then L3 fuses: e += LN(msg), agg[dst] += LN(msg) (atomics)
        hipLaunchKernelGGL(zero_f32_kernel, dim3((NN * L + 255) / 256), dim3(256), 0, stream,
                           agg, NN * L);
        GEMM(128, 1, 1, NE, L, nf, hid, nf, nf, nf, wt_be3[i], be_b3, be_g, be_bt, e, agg, dst, nhw, e);

        // NodeBlock: h += MLP([h, agg])
        GEMM(256, 3, 0, NN, 256, nf, nh, h, nf, agg, wt_bn1[i], bn_b1, nf, nf, nf, nfw, ni, hid, nfw);
        GEMM(128, 1, 0, NN, L, nf, hid, nf, nf, nf, wt_bn2[i], bn_b2, nf, nf, nf, nfw, ni, hid, nfw);
        GEMM(128, 1, 1, NN, L, nf, hid, nf, nf, nf, wt_bn3[i], bn_b3, bn_g, bn_bt, h, nfw, ni, nhw, h);
    }

    // ---- decoder ----
    GEMM(128, 0, 0, NN, L, h, nh, nf, nf, nf, wt_d1, F(36), nf, nf, nf, nfw, ni, hid, nfw);
    GEMM(128, 1, 0, NN, L, nf, hid, nf, nf, nf, wt_d2, F(38), nf, nf, nf, nfw, ni, hid, nfw);
    hipLaunchKernelGGL(dec_final_kernel, dim3((NN + 255) / 256), dim3(256), 0, stream,
                       hid, F(39), F(40), (float*)d_out, NN);
#undef GEMM
}
